// BitLinear_33191507264220
// MI455X (gfx1250) — 8-bit matrix operand order repaired and checked by reading, compiled, not run on hardware
//
#include <hip/hip_runtime.h>
#include <cstdint>
#include <cstddef>


#define M_ROWS 8192
#define IN_F   4096
#define OUT_F  16384
#define X_ELEMS (size_t(M_ROWS) * IN_F)

typedef __attribute__((ext_vector_type(16))) int   v16i;
typedef __attribute__((ext_vector_type(8)))  float v8f;

union Frag {
  v16i  v;
  uint4 q[4];
  uint2 d[8];
};

#if __has_builtin(__builtin_amdgcn_global_load_async_to_lds_b128)
#define USE_ASYNC_LDS 1
#else
#define USE_ASYNC_LDS 0
#endif

#if USE_ASYNC_LDS
typedef __attribute__((__vector_size__(4 * sizeof(int)))) int vs4i;
typedef __attribute__((address_space(1))) vs4i g_vs4i;
typedef __attribute__((address_space(3))) vs4i l_vs4i;

__device__ __forceinline__ g_vs4i* glob_v4(const void* p)
{
  return (g_vs4i*)(uint64_t)p;
}
__device__ __forceinline__ l_vs4i* lds_v4(const void* p)
{
  return (l_vs4i*)(uint32_t)(uint64_t)p;
}
#endif

__device__ __forceinline__ void wait_async_staging()
{
#if USE_ASYNC_LDS
#if __has_builtin(__builtin_amdgcn_s_wait_asynccnt)
  __builtin_amdgcn_s_wait_asynccnt(0);
#else
  asm volatile("s_wait_asynccnt 0" ::: "memory");
#endif
#endif
}

__global__ __launch_bounds__(256)
void reduce_absw_stage1(const float* __restrict__ W, float* __restrict__ partials)
{
  __shared__ float red[256];
  const float4* p = (const float4*)(W + (size_t)blockIdx.x * 16384);
  float s = 0.f;
#pragma unroll
  for (int j = 0; j < 16; ++j) {
    float4 v = p[threadIdx.x + j * 256];
    s += fabsf(v.x) + fabsf(v.y) + fabsf(v.z) + fabsf(v.w);
  }
  red[threadIdx.x] = s;
  __syncthreads();
  for (int off = 128; off > 0; off >>= 1) {
    if (threadIdx.x < (unsigned)off) red[threadIdx.x] += red[threadIdx.x + off];
    __syncthreads();
  }
  if (threadIdx.x == 0) partials[blockIdx.x] = red[0];
}

__global__ __launch_bounds__(256)
void reduce_absw_stage2(const float* __restrict__ partials, float* __restrict__ scaleW)
{
  __shared__ float red[256];
  float s = 0.f;
#pragma unroll
  for (int j = 0; j < 16; ++j) s += partials[threadIdx.x + j * 256];
  red[threadIdx.x] = s;
  __syncthreads();
  for (int off = 128; off > 0; off >>= 1) {
    if (threadIdx.x < (unsigned)off) red[threadIdx.x] += red[threadIdx.x + off];
    __syncthreads();
  }
  if (threadIdx.x == 0)
    scaleW[0] = fmaxf(red[0] * (1.0f / 67108864.0f), 1e-8f);
}

__device__ __forceinline__ uint32_t tern_fp8(float w, float t)
{
  return w > t ? 0x38u : (w < -t ? 0xB8u : 0x00u);
}

__global__ __launch_bounds__(256)
void quant_w_ternary_fp8(const float* __restrict__ W,
                         const float* __restrict__ scaleW,
                         uint32_t* __restrict__ W8)
{
  const float t = 0.5f * scaleW[0];
  const size_t base = ((size_t)blockIdx.x * 256 + threadIdx.x) * 4;
  const float4* src = (const float4*)W;
#pragma unroll
  for (int j = 0; j < 4; ++j) {
    float4 w = src[base + j];
    uint32_t b = tern_fp8(w.x, t) | (tern_fp8(w.y, t) << 8) |
                 (tern_fp8(w.z, t) << 16) | (tern_fp8(w.w, t) << 24);
    W8[base + j] = b;
  }
}

__global__ __launch_bounds__(256)
void quant_x_fp8_pair(const float* __restrict__ X,
                      uint32_t* __restrict__ Xhi,
                      uint32_t* __restrict__ Xlo)
{
  const size_t i = (size_t)blockIdx.x * 256 + threadIdx.x;
  float4 v = ((const float4*)X)[i];
  int h = __builtin_amdgcn_cvt_pk_fp8_f32(v.x, v.y, 0, false);
  h     = __builtin_amdgcn_cvt_pk_fp8_f32(v.z, v.w, h, true);
  float r0 = v.x - __builtin_amdgcn_cvt_f32_fp8(h, 0);
  float r1 = v.y - __builtin_amdgcn_cvt_f32_fp8(h, 1);
  float r2 = v.z - __builtin_amdgcn_cvt_f32_fp8(h, 2);
  float r3 = v.w - __builtin_amdgcn_cvt_f32_fp8(h, 3);
  int l = __builtin_amdgcn_cvt_pk_fp8_f32(r0, r1, 0, false);
  l     = __builtin_amdgcn_cvt_pk_fp8_f32(r2, r3, l, true);
  Xhi[i] = (uint32_t)h;
  Xlo[i] = (uint32_t)l;
}

#define LDS_PITCH 144
#define TILE_BYTES (128 * LDS_PITCH)

__global__ __launch_bounds__(256)
void bitlinear_gemm_fp8(const uint8_t* __restrict__ Ahi,
                        const uint8_t* __restrict__ Alo,
                        const uint8_t* __restrict__ W8,
                        const float*   __restrict__ scaleW,
                        float*         __restrict__ out)
{
  __shared__ uint8_t sAhi[2][TILE_BYTES];
  __shared__ uint8_t sAlo[2][TILE_BYTES];
  __shared__ uint8_t sB  [2][TILE_BYTES];

  const int tid  = threadIdx.x;
  const int lane = tid & 31;
  const int wave = tid >> 5;
  const int wm   = wave & 3;
  const int wn   = wave >> 2;
  const int lm   = lane & 15;
  const int lh   = lane >> 4;

  const int m0 = blockIdx.y * 128;
  const int n0 = blockIdx.x * 128;

  v8f zero = {};
  v8f acc[2][4];
#pragma unroll
  for (int i = 0; i < 2; ++i)
#pragma unroll
    for (int j = 0; j < 4; ++j) acc[i][j] = zero;

  const int lrow = tid >> 1;
  const int lcol = (tid & 1) * 64;
  const uint8_t* gah = Ahi + (size_t)(m0 + lrow) * IN_F + lcol;
  const uint8_t* gal = Alo + (size_t)(m0 + lrow) * IN_F + lcol;
  const uint8_t* gw  = W8  + (size_t)(n0 + lrow) * IN_F + lcol;
  const int ldst = lrow * LDS_PITCH + lcol;

  auto stage = [&](int buf, int k0) {
#if USE_ASYNC_LDS
    uint8_t* dah = &sAhi[buf][ldst];
    uint8_t* dal = &sAlo[buf][ldst];
    uint8_t* dw  = &sB  [buf][ldst];
#pragma unroll
    for (int j = 0; j < 4; ++j) {
      __builtin_amdgcn_global_load_async_to_lds_b128(
          glob_v4(gah + k0 + 16 * j), lds_v4(dah + 16 * j), 0, 0);
      __builtin_amdgcn_global_load_async_to_lds_b128(
          glob_v4(gal + k0 + 16 * j), lds_v4(dal + 16 * j), 0, 0);
      __builtin_amdgcn_global_load_async_to_lds_b128(
          glob_v4(gw + k0 + 16 * j), lds_v4(dw + 16 * j), 0, 0);
    }
#else
    uint4* dah = (uint4*)&sAhi[buf][ldst];
    uint4* dal = (uint4*)&sAlo[buf][ldst];
    uint4* dw  = (uint4*)&sB  [buf][ldst];
#pragma unroll
    for (int j = 0; j < 4; ++j) {
      dah[j] = *(const uint4*)(gah + k0 + 16 * j);
      dal[j] = *(const uint4*)(gal + k0 + 16 * j);
      dw [j] = *(const uint4*)(gw  + k0 + 16 * j);
    }
#endif
  };

  stage(0, 0);

  for (int k0 = 0; k0 < IN_F; k0 += 128) {
    const int cur = (k0 >> 7) & 1;

    wait_async_staging();
    __syncthreads();

    if (k0 + 128 < IN_F) stage(cur ^ 1, k0 + 128);

    Frag bfrag[4];
#pragma unroll
    for (int ni = 0; ni < 4; ++ni) {
      const uint8_t* bp = &sB[cur][(wn * 64 + ni * 16 + lm) * LDS_PITCH + 16 * lh];
#pragma unroll
      for (int j = 0; j < 4; ++j)
        bfrag[ni].q[j] = *(const uint4*)(bp + 32 * j);
    }

#pragma unroll
    for (int mi = 0; mi < 2; ++mi) {
      const int arow = (wm * 32 + mi * 16 + lm) * LDS_PITCH + 16 * lh;
      Frag ah, al;
#pragma unroll
      for (int j = 0; j < 8; ++j) {
        ah.d[j] = *(const uint2*)(&sAhi[cur][arow + 32 * (j >> 1) + 8 * (j & 1)]);
        al.d[j] = *(const uint2*)(&sAlo[cur][arow + 32 * (j >> 1) + 8 * (j & 1)]);
      }
#pragma unroll
      for (int ni = 0; ni < 4; ++ni) {
        acc[mi][ni] = __builtin_amdgcn_wmma_f32_16x16x128_fp8_fp8(
            ah.v, bfrag[ni].v, (short)0, acc[mi][ni], false, false);
        acc[mi][ni] = __builtin_amdgcn_wmma_f32_16x16x128_fp8_fp8(
            al.v, bfrag[ni].v, (short)0, acc[mi][ni], false, false);
      }
    }
  }

  const float s = scaleW[0];
#pragma unroll
  for (int mi = 0; mi < 2; ++mi) {
#pragma unroll
    for (int ni = 0; ni < 4; ++ni) {
      const int mbase = m0 + wm * 32 + mi * 16 + lh * 8;
      const int n     = n0 + wn * 64 + ni * 16 + lm;
      float* op = out + (size_t)mbase * OUT_F + n;
#pragma unroll
      for (int r = 0; r < 8; ++r)
        op[(size_t)r * OUT_F] = acc[mi][ni][r] * s;
    }
  }
}

extern "C" void kernel_launch(void* const* d_in, const int* in_sizes, int n_in,
                              void* d_out, int out_size, void* d_ws, size_t ws_size,
                              hipStream_t stream)
{
  const float* x = (const float*)d_in[0];
  const float* w = (const float*)d_in[1];
  float* out = (float*)d_out;

  char* ws = (char*)d_ws;
  float*   partials = (float*)ws;
  float*   scaleW   = (float*)(ws + (16u << 10));
  uint8_t* Xhi      = (uint8_t*)(ws + (128u << 10));
  uint8_t* Xlo      = Xhi + X_ELEMS;
  uint8_t* W8       = Xlo + X_ELEMS;

  hipLaunchKernelGGL(reduce_absw_stage1, dim3(4096), dim3(256), 0, stream, w, partials);
  hipLaunchKernelGGL(reduce_absw_stage2, dim3(1), dim3(256), 0, stream, partials, scaleW);
  hipLaunchKernelGGL(quant_w_ternary_fp8, dim3(16384), dim3(256), 0, stream,
                     w, scaleW, (uint32_t*)W8);
  hipLaunchKernelGGL(quant_x_fp8_pair, dim3(32768), dim3(256), 0, stream,
                     x, (uint32_t*)Xhi, (uint32_t*)Xlo);
  hipLaunchKernelGGL(bitlinear_gemm_fp8, dim3(OUT_F / 128, M_ROWS / 128), dim3(256),
                     0, stream, Xhi, Xlo, W8, scaleW, out);
}
